// CTLSTM_63900523430349
// MI455X (gfx1250) — matrix operand order repaired and checked by reading, not compiled, not run on hardware
//
#include <hip/hip_runtime.h>
#include <hip/hip_bf16.h>

typedef __attribute__((ext_vector_type(16))) _Float16 hv16;
typedef __attribute__((ext_vector_type(8)))  _Float16 hv8;
typedef __attribute__((ext_vector_type(8)))  float    fv8;

#define HDIM   1024
#define BATCH  32
#define TLEN   512
#define K2     2048   // 2*HDIM
#define N7     7168   // 7*HDIM

__device__ __forceinline__ float sigmoidf_(float x) { return 1.0f / (1.0f + __expf(-x)); }
__device__ __forceinline__ float softplusf_(float x) {
    // stable log1p(exp(x)) = max(x,0) + log1p(exp(-|x|))
    return fmaxf(x, 0.0f) + log1pf(__expf(-fabsf(x)));
}

// ---------------------------------------------------------------------------
// Repack rec_W (f32, row-major [2048][7168]) into f16 WMMA B-fragments.
// Fragment index: (((jg*7 + g)*64 + kt)*32 + lane)*16 + i
//   lane holds column n = g*1024 + jg*16 + (lane&15),
//   K values k = kt*32 + (lane>>4)*16 + i   (i = 0..15, contiguous)
// ---------------------------------------------------------------------------
__global__ __launch_bounds__(256) void repack_rec(const float* __restrict__ W,
                                                  _Float16* __restrict__ Wp) {
    size_t idx = (size_t)blockIdx.x * 256 + threadIdx.x;  // < 2048*7168
    int i    = (int)(idx & 15);
    int lane = (int)((idx >> 4) & 31);
    int kt   = (int)((idx >> 9) & 63);
    int g    = (int)((idx >> 15) % 7);
    int jg   = (int)(idx / (size_t)(16 * 32 * 64 * 7));
    int k = kt * 32 + (lane >> 4) * 16 + i;
    int n = g * HDIM + jg * 16 + (lane & 15);
    Wp[idx] = (_Float16)W[(size_t)k * N7 + n];
}

// Same repack for square [1024][1024] weights (w1_W, w2_W): 32 kt, 64 jg.
__global__ __launch_bounds__(256) void repack_sq(const float* __restrict__ W,
                                                 _Float16* __restrict__ Wp) {
    size_t idx = (size_t)blockIdx.x * 256 + threadIdx.x;  // < 1024*1024
    int i    = (int)(idx & 15);
    int lane = (int)((idx >> 4) & 31);
    int kt   = (int)((idx >> 9) & 31);
    int jg   = (int)(idx >> 14);
    int k = kt * 32 + (lane >> 4) * 16 + i;
    int n = jg * 16 + (lane & 15);
    Wp[idx] = (_Float16)W[(size_t)k * HDIM + n];
}

// X0 = [emb[0] broadcast, zeros]; c_d = 0; c_bar = 0.
__global__ __launch_bounds__(256) void init_state(const float* __restrict__ emb,
                                                  _Float16* __restrict__ X0,
                                                  float* __restrict__ c_d,
                                                  float* __restrict__ cbar) {
    int idx = blockIdx.x * 256 + threadIdx.x;  // < 32*1024
    int m = idx >> 10, j = idx & 1023;
    c_d[idx]  = 0.0f;
    cbar[idx] = 0.0f;
    X0[(size_t)m * K2 + j]        = (_Float16)emb[j];
    X0[(size_t)m * K2 + HDIM + j] = (_Float16)0.0f;
}

// ---------------------------------------------------------------------------
// One fused CT-LSTM step:
//   recurrence: g = Xin @ rec_W + rec_b, gates -> c, c_bar  (WMMA)
//   decay(t):   c_d' = c_bar + (c - c_bar)*exp(-delta*dur_t); h_d = o*tanh(c_d')
//   write Xout = [emb[event_t], h_d] (f16), update c_d / c_bar buffers.
// grid = (64 j-groups, 2 M-tiles); block = 1024 threads (32 waves).
// Waves 0..27: wave = gate(7) x K-quarter(4); each does 16x v_wmma f16.
// ---------------------------------------------------------------------------
__global__ __launch_bounds__(1024) void fused_step(
    const _Float16* __restrict__ Xin, _Float16* __restrict__ Xout,
    const _Float16* __restrict__ Wp, const float* __restrict__ rec_b,
    float* __restrict__ c_d, float* __restrict__ cbar,
    const float* __restrict__ time_seqs, const int* __restrict__ event_seqs,
    const float* __restrict__ emb, int t)
{
    __shared__ float red[4 * 7 * 256];  // [kq][gate][mm*16+jj]
    const int tid  = threadIdx.x;
    const int lane = tid & 31;
    const int wave = tid >> 5;
    const int jg    = blockIdx.x;  // hidden-unit group (16 units)
    const int mtile = blockIdx.y;  // batch tile (16 rows)

    if (wave < 28) {
        const int gate = wave % 7;
        const int kq   = wave / 7;          // K quarter: 16 kt-steps of 32
        const int lh = lane >> 4, lm = lane & 15;
        fv8 acc = {};
        const _Float16* wp = Wp + ((size_t)(jg * 7 + gate) * 64 + kq * 16) * 512 + lane * 16;
        const _Float16* xr = Xin + (size_t)(mtile * 16 + lm) * K2 + kq * 512 + lh * 16;
        for (int kt = 0; kt < 16; ++kt) {
            union { hv16 v; hv8 h[2]; } a, b;
            a.h[0] = *(const hv8*)(xr);        // K = k0 + lh*16 .. +8
            a.h[1] = *(const hv8*)(xr + 8);    // K = k0 + lh*16 + 8 .. +8
            b.v    = *(const hv16*)(wp);       // contiguous repacked fragment
            acc = __builtin_amdgcn_wmma_f32_16x16x32_f16(
                false, a.v, false, b.v, (short)0, acc, false, false);
            xr += 32;
            wp += 512;
        }
        float* l = red + (kq * 7 + gate) * 256;
#pragma unroll
        for (int r = 0; r < 8; ++r) l[(lh * 8 + r) * 16 + lm] = acc[r];
    }
    __syncthreads();

    if (tid < 256) {
        const int mm = tid >> 4, jj = tid & 15;
        const int m = mtile * 16 + mm;
        const int j = jg * 16 + jj;
        float g[7];
#pragma unroll
        for (int gg = 0; gg < 7; ++gg) {
            float s = rec_b[gg * HDIM + j];
#pragma unroll
            for (int kq = 0; kq < 4; ++kq) s += red[(kq * 7 + gg) * 256 + mm * 16 + jj];
            g[gg] = s;
        }
        // gate order: gi, gf, gz, go, gib, gfb, gd
        float gi  = sigmoidf_(g[0]);
        float gf  = sigmoidf_(g[1]);
        float gz  = tanhf(g[2]);
        float go  = sigmoidf_(g[3]);
        float gib = sigmoidf_(g[4]);
        float gfb = sigmoidf_(g[5]);
        float gd  = softplusf_(g[6]);

        const int o = m * HDIM + j;
        float cdv = c_d[o];
        float cbv = cbar[o];
        float c  = gf * cdv + gi * gz;       // c_tm1 = decayed c from prev step
        float cb = gfb * cbv + gib * gz;

        // decay phase for this scan step
        float dur = time_seqs[m * TLEN + t];
        float cdn = cb + (c - cb) * __expf(-gd * dur);
        float hd  = go * tanhf(cdn);
        c_d[o]  = cdn;
        cbar[o] = cb;

        int ev = event_seqs[m * TLEN + t];
        Xout[(size_t)m * K2 + j]        = (_Float16)emb[(size_t)ev * HDIM + j];
        Xout[(size_t)m * K2 + HDIM + j] = (_Float16)hd;
    }
}

// ---------------------------------------------------------------------------
// Small WMMA GEMM for the output MLP: Out[32xN] = act(A[32xK] @ Wp + bias)
// K = 1024 (32 kt), N = 1024. grid = (64, 2); block = 256 (8 waves split K).
// ---------------------------------------------------------------------------
template <bool RELU, bool OUTF16>
__global__ __launch_bounds__(256) void mlp_gemm(
    const _Float16* __restrict__ A, int lda,
    const _Float16* __restrict__ Wp, const float* __restrict__ bias,
    void* __restrict__ Out, int ldo)
{
    __shared__ float red[8 * 256];
    const int tid  = threadIdx.x;
    const int lane = tid & 31;
    const int wave = tid >> 5;       // K split: 4 kt per wave
    const int jg    = blockIdx.x;
    const int mtile = blockIdx.y;
    const int lh = lane >> 4, lm = lane & 15;

    fv8 acc = {};
    const _Float16* wp = Wp + ((size_t)jg * 32 + wave * 4) * 512 + lane * 16;
    const _Float16* xr = A + (size_t)(mtile * 16 + lm) * lda + wave * 128 + lh * 16;
    for (int kt = 0; kt < 4; ++kt) {
        union { hv16 v; hv8 h[2]; } a, b;
        a.h[0] = *(const hv8*)(xr);
        a.h[1] = *(const hv8*)(xr + 8);
        b.v    = *(const hv16*)(wp);
        acc = __builtin_amdgcn_wmma_f32_16x16x32_f16(
            false, a.v, false, b.v, (short)0, acc, false, false);
        xr += 32;
        wp += 512;
    }
    float* l = red + wave * 256;
#pragma unroll
    for (int r = 0; r < 8; ++r) l[(lh * 8 + r) * 16 + lm] = acc[r];
    __syncthreads();

    const int mm = tid >> 4, jj = tid & 15;
    const int m = mtile * 16 + mm;
    const int j = jg * 16 + jj;
    float s = bias[j];
#pragma unroll
    for (int w = 0; w < 8; ++w) s += red[w * 256 + mm * 16 + jj];
    if (RELU) s = fmaxf(s, 0.0f);
    if (OUTF16) ((_Float16*)Out)[(size_t)m * ldo + j] = (_Float16)s;
    else        ((float*)Out)[(size_t)m * ldo + j]    = s;
}

// ---------------------------------------------------------------------------
extern "C" void kernel_launch(void* const* d_in, const int* in_sizes, int n_in,
                              void* d_out, int out_size, void* d_ws, size_t ws_size,
                              hipStream_t stream) {
    (void)in_sizes; (void)n_in; (void)out_size; (void)ws_size;
    const int*   ev   = (const int*)  d_in[0];  // event_seqs  (B,T) int32
    const float* ts   = (const float*)d_in[1];  // time_seqs   (B,T)
    const float* recW = (const float*)d_in[2];  // rec_W (2H,7H)
    const float* recb = (const float*)d_in[3];  // rec_b (7H,)
    const float* emb  = (const float*)d_in[4];  // emb (33,H)
    const float* w1W  = (const float*)d_in[5];
    const float* w1b  = (const float*)d_in[6];
    const float* w2W  = (const float*)d_in[7];
    const float* w2b  = (const float*)d_in[8];

    char* ws = (char*)d_ws;
    size_t off = 0;
    auto take = [&](size_t bytes) -> char* {
        char* p = ws + off;
        off += (bytes + 255) & ~(size_t)255;
        return p;
    };
    _Float16* Wp  = (_Float16*)take((size_t)K2 * N7 * 2);        // 29.4 MB
    _Float16* W1p = (_Float16*)take((size_t)HDIM * HDIM * 2);    // 2 MB
    _Float16* W2p = (_Float16*)take((size_t)HDIM * HDIM * 2);    // 2 MB
    _Float16* X0  = (_Float16*)take((size_t)BATCH * K2 * 2);
    _Float16* X1  = (_Float16*)take((size_t)BATCH * K2 * 2);
    float*    cd  = (float*)   take((size_t)BATCH * HDIM * 4);
    float*    cb  = (float*)   take((size_t)BATCH * HDIM * 4);
    _Float16* hid = (_Float16*)take((size_t)BATCH * HDIM * 2);

    // One-time weight repack (f32 -> f16 WMMA fragments) + state init.
    repack_rec<<<(K2 * N7) / 256, 256, 0, stream>>>(recW, Wp);
    repack_sq <<<(HDIM * HDIM) / 256, 256, 0, stream>>>(w1W, W1p);
    repack_sq <<<(HDIM * HDIM) / 256, 256, 0, stream>>>(w2W, W2p);
    init_state<<<(BATCH * HDIM) / 256, 256, 0, stream>>>(emb, X0, cd, cb);

    // 512 chained fused steps (sequential recurrence), ping-pong X buffers.
    for (int t = 0; t < TLEN; ++t) {
        const _Float16* xin = (t & 1) ? X1 : X0;
        _Float16*       xout = (t & 1) ? X0 : X1;
        fused_step<<<dim3(HDIM / 16, 2), 1024, 0, stream>>>(
            xin, xout, Wp, recb, cd, cb, ts, ev, emb, t);
    }
    // t=511 wrote X0; h_last = X0[:, H:2H] (f16, row stride 2048).
    mlp_gemm<true, true ><<<dim3(HDIM / 16, 2), 256, 0, stream>>>(
        X0 + HDIM, K2, W1p, w1b, hid, HDIM);
    mlp_gemm<false, false><<<dim3(HDIM / 16, 2), 256, 0, stream>>>(
        hid, HDIM, W2p, w2b, d_out, HDIM);
}
